// MiniYolo_50156628082694
// MI455X (gfx1250) — matrix operand order repaired and checked by reading, not compiled, not run on hardware
//
#include <hip/hip_runtime.h>

typedef __attribute__((ext_vector_type(16))) _Float16 v16h;
typedef __attribute__((ext_vector_type(8)))  float    v8f;
typedef __attribute__((ext_vector_type(4)))  unsigned int u32x4;
typedef __attribute__((ext_vector_type(8)))  int i32x8;
typedef __attribute__((ext_vector_type(4)))  int i32x4;

#define BN_EPS 1e-5f

// ---------------------------------------------------------------------------
// Prep: fold BN into conv weights, convert f16, and write directly in WMMA
// B-fragment order: wfrag[kt][oct][lane][e], where element e of lane l is
// K = kt*32 + (l>=16 ? 8 : 0) + (e&7) + (e>=8 ? 16 : 0), N(oc) = oct*16 + (l&15).
// K ordering (channel-minor im2col): K = (ky*3+kx)*CINP + c.
// ---------------------------------------------------------------------------
__global__ void prep_cbn_kernel(const float* __restrict__ w,
                                const float* __restrict__ g,
                                const float* __restrict__ bb,
                                const float* __restrict__ mm,
                                const float* __restrict__ vv,
                                int CIN, int CINP, int COUT, int KP,
                                _Float16* __restrict__ wfrag,
                                float* __restrict__ bias) {
  int idx = blockIdx.x * blockDim.x + threadIdx.x;
  if (idx >= KP * COUT) return;
  int e    = idx & 15;
  int lane = (idx >> 4) & 31;
  int rest = idx >> 9;
  int NOCT = COUT >> 4;
  int oct  = rest % NOCT;
  int kt   = rest / NOCT;
  int K  = kt * 32 + ((lane >> 4) * 8) + (e & 7) + ((e & 8) << 1);
  int oc = oct * 16 + (lane & 15);
  float scale = g[oc] * rsqrtf(vv[oc] + BN_EPS);
  int c = K % CINP;   // channel-minor
  int s = K / CINP;   // ky*3+kx
  float val = 0.0f;
  if (s < 9 && c < CIN) val = w[(oc * CIN + c) * 9 + s] * scale;
  wfrag[idx] = (_Float16)val;
  if (K == 0) bias[oc] = bb[oc] - mm[oc] * scale;
}

// 1x1 end conv: wend (125,128) f32 -> fragment order, oc padded to 128
__global__ void prep_end_kernel(const float* __restrict__ wend,
                                const float* __restrict__ bend,
                                _Float16* __restrict__ wfrag,
                                float* __restrict__ bias) {
  int idx = blockIdx.x * blockDim.x + threadIdx.x;
  if (idx >= 128 * 128) return;
  int e    = idx & 15;
  int lane = (idx >> 4) & 31;
  int rest = idx >> 9;
  int oct  = rest & 7;
  int kt   = rest >> 3;
  int K  = kt * 32 + ((lane >> 4) * 8) + (e & 7) + ((e & 8) << 1);
  int oc = oct * 16 + (lane & 15);
  wfrag[idx] = (oc < 125) ? (_Float16)wend[oc * 128 + K] : (_Float16)0.0f;
  if (K == 0) bias[oc] = (oc < 125) ? bend[oc] : 0.0f;
}

// ---------------------------------------------------------------------------
// Permute(0,3,2,1) + bilinear resize (align_corners) 640 -> 416, f32 -> f16
// Output (B,416,416,4) NHWC, channel 3 zero-padded.
// ---------------------------------------------------------------------------
__global__ void resize_kernel(const float* __restrict__ x,
                              _Float16* __restrict__ out) {
  size_t idx = (size_t)blockIdx.x * blockDim.x + threadIdx.x;
  const size_t total = (size_t)16 * 416 * 416;
  if (idx >= total) return;
  int j = (int)(idx % 416);
  size_t t = idx / 416;
  int i = (int)(t % 416);
  int b = (int)(t / 416);
  const float s = 639.0f / 415.0f;
  float fw = i * s;
  float fh = j * s;
  int w0 = (int)floorf(fw); int h0 = (int)floorf(fh);
  int w1 = (w0 + 1 > 639) ? 639 : w0 + 1;
  int h1 = (h0 + 1 > 639) ? 639 : h0 + 1;
  float aw = fw - (float)w0, ah = fh - (float)h0;
  const float* p = x + (size_t)b * 640 * 640 * 3;
  _Float16* o = out + idx * 4;
#pragma unroll
  for (int c = 0; c < 3; ++c) {
    float v00 = p[((size_t)h0 * 640 + w0) * 3 + c];
    float v01 = p[((size_t)h0 * 640 + w1) * 3 + c];
    float v10 = p[((size_t)h1 * 640 + w0) * 3 + c];
    float v11 = p[((size_t)h1 * 640 + w1) * 3 + c];
    float r0 = v00 * (1.0f - aw) + v01 * aw;
    float r1 = v10 * (1.0f - aw) + v11 * aw;
    o[c] = (_Float16)(r0 * (1.0f - ah) + r1 * ah);
  }
  o[3] = (_Float16)0.0f;
}

// ---------------------------------------------------------------------------
// 3x3 conv (pad 1) + folded-BN bias + LeakyReLU(0.01), implicit GEMM.
// Block: 256 thr = 8 waves; tile = 64 px (one output row) x (NT*16) oc.
// Wave (w&3) -> 16-px M tile, (w>>2) -> NJ oc tiles (NJ accumulators).
// Slab: 3 rows x 66 cols x CINP f16 in LDS covering [y-1..y+1]x[x0-1..x0+64].
// USE_TDM: stage slab rows with the Tensor Data Mover (one 1-D descriptor per
// halo row, issued by waves 0..2); TDM OOB reads return zero, which handles
// the right halo; top/bottom/left edges pre-zero the slab.
// ---------------------------------------------------------------------------
template <int CINP, int COUT, int H, int KTOT, bool USE_TDM>
__global__ __launch_bounds__(256) void conv3x3_wmma(
    const _Float16* __restrict__ in, const _Float16* __restrict__ wfrag,
    const float* __restrict__ bias, _Float16* __restrict__ out) {
  constexpr int W = H;
  constexpr int TILES_X = (W + 63) / 64;
  constexpr int KP = ((KTOT + 31) / 32) * 32;
  constexpr int NOCT = COUT / 16;
  constexpr int NJ = (COUT >= 64) ? 2 : 1;
  constexpr int NT = 2 * NJ;
  constexpr int SLABN = 3 * 66 * CINP;
  constexpr int LOG2C = (CINP == 128) ? 7 : (CINP == 32) ? 5 : 2;

  __shared__ _Float16 slab[SLABN];

  int bid = blockIdx.x;
  int tx = bid % TILES_X;
  int rest = bid / TILES_X;
  int y = rest % H;
  int b = rest / H;
  int x0 = tx * 64;

  int tid = threadIdx.x, lane = tid & 31, wave = tid >> 5;
  int hi = lane >> 4, laneN = lane & 15;
  int mloc = (wave & 3) * 16 + laneN;
  int oct0 = blockIdx.y * NT + (wave >> 2) * NJ;

  if constexpr (USE_TDM) {
    bool edge = (y == 0) || (y == H - 1) || (x0 == 0);
    if (edge) {
      uint4 z = {0u, 0u, 0u, 0u};
      for (int i = tid; i < SLABN / 8; i += 256) ((uint4*)slab)[i] = z;
    }
    __syncthreads();
    if (wave < 3) {
      int yy = y + wave - 1;
      if (yy >= 0 && yy < H) {
        int cstart  = (x0 == 0) ? 0 : (x0 - 1);
        int colskip = (x0 == 0) ? 1 : 0;
        unsigned lds_off = (unsigned)(uintptr_t)(&slab[0]) +
                           (unsigned)((wave * 66 + colskip) * CINP) * 2u;
        const _Float16* gp = in + (((size_t)b * H + yy) * W + cstart) * CINP;
        unsigned long long ga = (unsigned long long)(uintptr_t)gp;
        unsigned td0   = (unsigned)((W - cstart) * CINP);   // tensor_dim0 (elems)
        unsigned tile0 = (unsigned)((66 - colskip) * CINP); // tile_dim0 (elems)
        unsigned long long st0 = (unsigned long long)W * CINP;
        // D# group0: count=1 | lds_addr | global_addr | type=2
        u32x4 g0;
        g0[0] = 1u;
        g0[1] = lds_off;
        g0[2] = (unsigned)(ga & 0xFFFFFFFFull);
        g0[3] = (unsigned)((ga >> 32) & 0x01FFFFFFull) | (2u << 30);
        // D# group1: data_size=1(2B); tensor_dim0/1; tile_dim0/1; dim0 stride
        i32x8 g1;
        g1[0] = (int)(1u << 16);
        g1[1] = (int)((td0 & 0xFFFFu) << 16);
        g1[2] = (int)((td0 >> 16) | (1u << 16));   // tensor_dim1 = 1
        g1[3] = (int)(tile0 << 16);
        g1[4] = (int)1;                            // tile_dim1=1, tile_dim2=0
        g1[5] = (int)(st0 & 0xFFFFFFFFull);
        g1[6] = (int)((st0 >> 32) & 0xFFFFull);
        g1[7] = 0;
        i32x4 gz = {0, 0, 0, 0};
#if defined(__clang_major__) && (__clang_major__ >= 23)
        i32x8 gz8 = {0, 0, 0, 0, 0, 0, 0, 0};
        __builtin_amdgcn_tensor_load_to_lds(g0, g1, gz, gz, gz8, 0);
#else
        __builtin_amdgcn_tensor_load_to_lds(g0, g1, gz, gz, 0);
#endif
      }
      __builtin_amdgcn_s_wait_tensorcnt((short)0);
    }
    __syncthreads();
  } else {
    // cooperative load (CINP==4 layer only): 4B chunks, border zero-padded
    constexpr int VPP = CINP / 2;
    constexpr int NV = 3 * 66 * VPP;
    for (int i = tid; i < NV; i += 256) {
      int cg = i % VPP;
      int p = i / VPP;
      int col = p % 66;
      int row = p / 66;
      int yy = y + row - 1;
      int xx = x0 + col - 1;
      unsigned int v = 0u;
      if (yy >= 0 && yy < H && xx >= 0 && xx < W)
        v = *(const unsigned int*)(in + (((size_t)b * H + yy) * W + xx) * CINP + cg * 2);
      *(unsigned int*)(slab + ((size_t)row * 66 + col) * CINP + cg * 2) = v;
    }
    __syncthreads();
  }

  v8f acc[NJ];
#pragma unroll
  for (int j = 0; j < NJ; ++j) acc[j] = (v8f){};

  for (int ks = 0; ks < KP; ks += 32) {
    // ---- A fragment: lane<16 K {0..7,16..23}+ks, lane>=16 +8 ----
    v16h a;
    if constexpr (CINP >= 8) {
      // channel-minor K: 8 consecutive K = 8 consecutive channels -> b128 LDS
      int b0k = ks + hi * 8;
      int s0 = b0k >> LOG2C; int c0 = b0k & (CINP - 1);
      int ky0 = (s0 * 11) >> 5; int kx0 = s0 - 3 * ky0;
      ((uint4*)&a)[0] = *(const uint4*)(slab + (ky0 * 66 + mloc + kx0) * CINP + c0);
      int b1k = b0k + 16;
      int s1 = b1k >> LOG2C; int c1 = b1k & (CINP - 1);
      int ky1 = (s1 * 11) >> 5; int kx1 = s1 - 3 * ky1;
      ((uint4*)&a)[1] = *(const uint4*)(slab + (ky1 * 66 + mloc + kx1) * CINP + c1);
    } else {
#pragma unroll
      for (int e = 0; e < 16; ++e) {
        int k = ks + (e & 7) + ((e & 8) << 1) + hi * 8;
        _Float16 v = (_Float16)0.0f;
        if (k < KTOT) {
          int c = k & (CINP - 1); int s = k >> LOG2C;
          int ky = (s * 11) >> 5; int kx = s - 3 * ky;
          v = slab[(ky * 66 + mloc + kx) * CINP + c];
        }
        a[e] = v;
      }
    }

    int kt = ks >> 5;
    if (ks + 32 < KP)
      __builtin_prefetch(wfrag + (((size_t)(kt + 1) * NOCT + oct0) * 32 + lane) * 16, 0, 1);

    // ---- B fragments from pre-swizzled weights: 2x b128 per tile ----
#pragma unroll
    for (int j = 0; j < NJ; ++j) {
      const _Float16* wl = wfrag + (((size_t)kt * NOCT + (oct0 + j)) * 32 + lane) * 16;
      v16h bf;
      ((uint4*)&bf)[0] = ((const uint4*)wl)[0];
      ((uint4*)&bf)[1] = ((const uint4*)wl)[1];
      acc[j] = __builtin_amdgcn_wmma_f32_16x16x32_f16(false, a, false, bf,
                                                      (short)0, acc[j], false, false);
    }
  }

  // ---- epilogue: bias + LeakyReLU, f16 NHWC store ----
#pragma unroll
  for (int j = 0; j < NJ; ++j) {
    int ocn = (oct0 + j) * 16 + laneN;
    float bv = bias[ocn];
#pragma unroll
    for (int r = 0; r < 8; ++r) {
      int x = x0 + (wave & 3) * 16 + r + hi * 8;
      if (x < W) {
        float v = acc[j][r] + bv;
        v = fmaxf(v, 0.01f * v);
        out[(((size_t)b * H + y) * W + x) * COUT + ocn] = (_Float16)v;
      }
    }
  }
}

// ---------------------------------------------------------------------------
// 2x2 max pool, stride 2, NHWC f16
// ---------------------------------------------------------------------------
template <int C, int HIN>
__global__ void pool2_kernel(const _Float16* __restrict__ in,
                             _Float16* __restrict__ out) {
  constexpr int HO = HIN / 2;
  const size_t total = (size_t)16 * HO * HO * C;
  size_t idx = (size_t)blockIdx.x * blockDim.x + threadIdx.x;
  if (idx >= total) return;
  int c = (int)(idx % C);
  size_t t = idx / C;
  int x = (int)(t % HO); t /= HO;
  int y = (int)(t % HO);
  int b = (int)(t / HO);
  const _Float16* p = in + (((size_t)b * HIN + 2 * y) * HIN + 2 * x) * C + c;
  float v0 = (float)p[0];
  float v1 = (float)p[C];
  float v2 = (float)p[(size_t)HIN * C];
  float v3 = (float)p[(size_t)HIN * C + C];
  out[idx] = (_Float16)fmaxf(fmaxf(v0, v1), fmaxf(v2, v3));
}

// ---------------------------------------------------------------------------
// 1x1 end conv as GEMM: M = 2704 px, K = 128, N = 128 (125 live), f32 out.
// ---------------------------------------------------------------------------
__global__ __launch_bounds__(256) void convend_wmma(
    const _Float16* __restrict__ in, const _Float16* __restrict__ wfrag,
    const float* __restrict__ bias, float* __restrict__ out) {
  int tid = threadIdx.x;
  int lane = tid & 31;
  int wave = tid >> 5;
  int hi = lane >> 4;
  int laneN = lane & 15;
  int m0 = blockIdx.x * 16;
  int oc = wave * 16 + laneN;

  const _Float16* arow = in + (size_t)(m0 + (lane & 15)) * 128;
  v8f acc = {};
#pragma unroll
  for (int ks = 0; ks < 128; ks += 32) {
    v16h a;
    *((uint4*)&a)       = *(const uint4*)(arow + ks + hi * 8);
    *(((uint4*)&a) + 1) = *(const uint4*)(arow + ks + 16 + hi * 8);
    const _Float16* wl = wfrag + (((size_t)(ks >> 5) * 8 + wave) * 32 + lane) * 16;
    v16h bf;
    ((uint4*)&bf)[0] = ((const uint4*)wl)[0];
    ((uint4*)&bf)[1] = ((const uint4*)wl)[1];
    acc = __builtin_amdgcn_wmma_f32_16x16x32_f16(false, a, false, bf,
                                                 (short)0, acc, false, false);
  }
  float bv = bias[oc];
#pragma unroll
  for (int r = 0; r < 8; ++r) {
    int m = m0 + r + hi * 8;
    if (oc < 125) out[(size_t)m * 125 + oc] = acc[r] + bv;
  }
}

// ---------------------------------------------------------------------------
// host launcher
// ---------------------------------------------------------------------------
extern "C" void kernel_launch(void* const* d_in, const int* in_sizes, int n_in,
                              void* d_out, int out_size, void* d_ws, size_t ws_size,
                              hipStream_t stream) {
  const float* x = (const float*)d_in[0];
  const float *Wt[6], *G[6], *Bb[6], *Mn[6], *Vr[6];
  for (int i = 0; i < 6; ++i) {
    Wt[i] = (const float*)d_in[1 + 5 * i + 0];
    G[i]  = (const float*)d_in[1 + 5 * i + 1];
    Bb[i] = (const float*)d_in[1 + 5 * i + 2];
    Mn[i] = (const float*)d_in[1 + 5 * i + 3];
    Vr[i] = (const float*)d_in[1 + 5 * i + 4];
  }
  const float* wend = (const float*)d_in[31];
  const float* bend = (const float*)d_in[32];
  float* out = (float*)d_out;

  char* ws = (char*)d_ws;
  size_t off = 0;
  auto alloc = [&](size_t bytes) -> void* {
    void* p = ws + off;
    off += bytes;
    off = (off + 255) & ~(size_t)255;
    return p;
  };

  const int KPs[6]   = {64, 288, 1152, 1152, 1152, 1152};
  const int CINs[6]  = {3, 32, 128, 128, 128, 128};
  const int CINPs[6] = {4, 32, 128, 128, 128, 128};
  const int COUTs[6] = {32, 128, 128, 128, 128, 128};
  _Float16* w16[6];
  float* bias[6];
  for (int i = 0; i < 6; ++i) {
    w16[i]  = (_Float16*)alloc((size_t)KPs[i] * COUTs[i] * sizeof(_Float16));
    bias[i] = (float*)alloc((size_t)COUTs[i] * sizeof(float));
  }
  _Float16* w16e = (_Float16*)alloc((size_t)128 * 128 * sizeof(_Float16));
  float* biase = (float*)alloc(128 * sizeof(float));
  _Float16* regA = (_Float16*)alloc((size_t)23000000 * sizeof(_Float16)); // max 22.16M elems
  _Float16* regB = (_Float16*)alloc((size_t)89000000 * sizeof(_Float16)); // max 88.61M elems

  for (int i = 0; i < 6; ++i) {
    int total = KPs[i] * COUTs[i];
    prep_cbn_kernel<<<(total + 255) / 256, 256, 0, stream>>>(
        Wt[i], G[i], Bb[i], Mn[i], Vr[i], CINs[i], CINPs[i], COUTs[i], KPs[i],
        w16[i], bias[i]);
  }
  prep_end_kernel<<<(128 * 128 + 255) / 256, 256, 0, stream>>>(wend, bend, w16e, biase);

  resize_kernel<<<(int)(((size_t)16 * 416 * 416 + 255) / 256), 256, 0, stream>>>(x, regA);

  // cbn0 @416 (3->32), pool -> 208
  conv3x3_wmma<4, 32, 416, 36, false>
      <<<dim3(7 * 416 * 16, 1), 256, 0, stream>>>(regA, w16[0], bias[0], regB);
  pool2_kernel<32, 416>
      <<<(int)(((size_t)16 * 208 * 208 * 32 + 255) / 256), 256, 0, stream>>>(regB, regA);

  // cbn1 @208 (32->128), pool -> 104
  conv3x3_wmma<32, 128, 208, 288, true>
      <<<dim3(4 * 208 * 16, 2), 256, 0, stream>>>(regA, w16[1], bias[1], regB);
  pool2_kernel<128, 208>
      <<<(int)(((size_t)16 * 104 * 104 * 128 + 255) / 256), 256, 0, stream>>>(regB, regA);

  // cbn2 @104, pool -> 52
  conv3x3_wmma<128, 128, 104, 1152, true>
      <<<dim3(2 * 104 * 16, 2), 256, 0, stream>>>(regA, w16[2], bias[2], regB);
  pool2_kernel<128, 104>
      <<<(int)(((size_t)16 * 52 * 52 * 128 + 255) / 256), 256, 0, stream>>>(regB, regA);

  // cbn3 @52, pool -> 26
  conv3x3_wmma<128, 128, 52, 1152, true>
      <<<dim3(1 * 52 * 16, 2), 256, 0, stream>>>(regA, w16[3], bias[3], regB);
  pool2_kernel<128, 52>
      <<<(int)(((size_t)16 * 26 * 26 * 128 + 255) / 256), 256, 0, stream>>>(regB, regA);

  // cbn4 @26, pool -> 13
  conv3x3_wmma<128, 128, 26, 1152, true>
      <<<dim3(1 * 26 * 16, 2), 256, 0, stream>>>(regA, w16[4], bias[4], regB);
  pool2_kernel<128, 26>
      <<<(int)(((size_t)16 * 13 * 13 * 128 + 255) / 256), 256, 0, stream>>>(regB, regA);

  // cbn5 @13 (no pool)
  conv3x3_wmma<128, 128, 13, 1152, true>
      <<<dim3(1 * 13 * 16, 2), 256, 0, stream>>>(regA, w16[5], bias[5], regB);

  // 1x1 end conv -> d_out (B,13,13,125) f32; second tuple output is empty.
  convend_wmma<<<169, 256, 0, stream>>>(regB, w16e, biase, out);
}
